// NTXentLoss_49220325212843
// MI455X (gfx1250) — matrix operand order repaired and checked by reading, not compiled, not run on hardware
//
#include <hip/hip_runtime.h>
#include <math.h>

// ---------------------------------------------------------------------------
// NT-Xent loss for N=8192, D=512, T=0.1 on gfx1250 (MI455X).
// loss = (1/N) * sum_j [ log(sum_{i!=j} exp(s_ij/T)) - s_{j,p(j)}/T ],
// p(j) = (j + 4096) mod 8192.  S = Xn Xn^T via split-bf16 WMMA; symmetry
// exploited: only upper-triangular 128x128 block tiles are computed, each
// producing row-partials AND column-partials of exp(10*s).
// ---------------------------------------------------------------------------

typedef __attribute__((ext_vector_type(16))) __bf16 v16bf;
typedef __attribute__((ext_vector_type(8)))  float  v8f;

#define NROWS 8192
#define DDIM  512
#define INV_T 10.0f
#define NBLK  64                  // 8192 / 128
#define NTRI  (NBLK * (NBLK + 1) / 2)   // 2080 upper-triangular block tiles

union Frag {
    int4  q[2];   // two b128 loads
    v16bf v;      // 16 bf16 elements for WMMA operand
};

// ---------------------------------------------------------------------------
// Kernel 1: row-normalize and split into bf16 hi/lo.  One block per row.
// ---------------------------------------------------------------------------
__global__ __launch_bounds__(256) void rownorm_split_kernel(
    const float* __restrict__ x, __bf16* __restrict__ xh, __bf16* __restrict__ xl)
{
    const int row = blockIdx.x;
    const int t   = threadIdx.x;                // 256 threads, 2 elems each
    const float* xr = x + (size_t)row * DDIM;

    float a0 = xr[t];
    float a1 = xr[t + 256];

    __shared__ float red[256];
    red[t] = a0 * a0 + a1 * a1;
    __syncthreads();
    #pragma unroll
    for (int off = 128; off > 0; off >>= 1) {
        if (t < off) red[t] += red[t + off];
        __syncthreads();
    }
    const float rn = 1.0f / fmaxf(sqrtf(red[0]), 1e-8f);

    const float n0 = a0 * rn, n1 = a1 * rn;
    const __bf16 h0 = (__bf16)n0;
    const __bf16 h1 = (__bf16)n1;
    const __bf16 l0 = (__bf16)(n0 - (float)h0);
    const __bf16 l1 = (__bf16)(n1 - (float)h1);

    const size_t base = (size_t)row * DDIM;
    xh[base + t]        = h0;
    xh[base + t + 256]  = h1;
    xl[base + t]        = l0;
    xl[base + t + 256]  = l1;
}

// ---------------------------------------------------------------------------
// Kernel 2: upper-triangular tiled split-bf16 WMMA GEMM.
// Each 256-thread block owns one 128x128 tile (br <= bc); 8 waves, one
// 16x128 strip each.  Emits:
//   row-partials  -> partial[row*64 + bc]   (sum over the tile's 128 cols)
//   col-partials  -> partial[col*64 + br]   (sum over the tile's 128 rows,
//                                            off-diagonal tiles only)
// ---------------------------------------------------------------------------
__global__ __launch_bounds__(256) void simtile_kernel(
    const __bf16* __restrict__ xh, const __bf16* __restrict__ xl,
    float* __restrict__ partial /* [NROWS][64] */)
{
    // Decode linear tile index -> (br, bc) with bc >= br.
    int t_ = blockIdx.x, br = 0;
    while (t_ >= NBLK - br) { t_ -= NBLK - br; ++br; }
    const int bc = br + t_;

    const int lane = threadIdx.x & 31;
    const int wave = threadIdx.x >> 5;       // 0..7
    const int lm   = lane & 15;              // 0..15
    const int hi   = lane >> 4;              // 0 or 1 (lane half)

    const int rowBase = br * 128 + wave * 16;
    const int colBase = bc * 128;

    v8f acc[8] = {};                          // 8 col-tiles of 16x16 f32

    // A-fragment addressing (16-bit A 16x32, same k order as B):
    //   lanes 0-15: K = k0+{0..15};  lanes 16-31: K = k0+16+{0..15}.
    const size_t aRow = (size_t)(rowBase + lm) * DDIM;
    // B-fragment addressing (16-bit B 32x16):
    //   lanes 0-15: K = k0+{0..15};  lanes 16-31: K = k0+16+{0..15}.
    for (int kb = 0; kb < 16; ++kb) {
        const int k0 = kb * 32;

        Frag Ah, Al;
        {
            const __bf16* pa = xh + aRow + k0 + hi * 16;
            Ah.q[0] = *(const int4*)(pa);
            Ah.q[1] = *(const int4*)(pa + 8);
            const __bf16* pb = xl + aRow + k0 + hi * 16;
            Al.q[0] = *(const int4*)(pb);
            Al.q[1] = *(const int4*)(pb + 8);
        }

        #pragma unroll
        for (int ct = 0; ct < 8; ++ct) {
            const size_t bRow = (size_t)(colBase + ct * 16 + lm) * DDIM;
            Frag Bh, Bl;
            const __bf16* pbh = xh + bRow + k0 + hi * 16;
            Bh.q[0] = *(const int4*)(pbh);
            Bh.q[1] = *(const int4*)(pbh + 8);
            const __bf16* pbl = xl + bRow + k0 + hi * 16;
            Bl.q[0] = *(const int4*)(pbl);
            Bl.q[1] = *(const int4*)(pbl + 8);

            // S ~= hi*hi + hi*lo + lo*hi  (lo*lo term ~2^-18, negligible)
            acc[ct] = __builtin_amdgcn_wmma_f32_16x16x32_bf16(
                false, Ah.v, false, Bh.v, (short)0, acc[ct], false, false);
            acc[ct] = __builtin_amdgcn_wmma_f32_16x16x32_bf16(
                false, Ah.v, false, Bl.v, (short)0, acc[ct], false, false);
            acc[ct] = __builtin_amdgcn_wmma_f32_16x16x32_bf16(
                false, Al.v, false, Bh.v, (short)0, acc[ct], false, false);
        }
    }

    // exp(10*s), mask exact diagonal, accumulate row- and column-partials.
    // C/D layout: elem r -> row = rowBase + r + hi*8, col = colBase + ct*16 + lm.
    float rowpart[8];
    float colpart[8];
    #pragma unroll
    for (int r = 0; r < 8; ++r) { rowpart[r] = 0.0f; colpart[r] = 0.0f; }

    #pragma unroll
    for (int ct = 0; ct < 8; ++ct) {
        const int col = colBase + ct * 16 + lm;
        float csum = 0.0f;
        #pragma unroll
        for (int r = 0; r < 8; ++r) {
            const int row = rowBase + r + hi * 8;
            float e = __expf(acc[ct][r] * INV_T);
            e = (row == col) ? 0.0f : e;     // only fires on diagonal tiles
            rowpart[r] += e;
            csum       += e;
        }
        colpart[ct] = csum;
    }

    // Row-partials: reduce across the 16 lanes of each half (same rows/half).
    #pragma unroll
    for (int off = 1; off < 16; off <<= 1) {
        #pragma unroll
        for (int r = 0; r < 8; ++r)
            rowpart[r] += __shfl_xor(rowpart[r], off, 16);
    }
    if (lm == 0) {
        #pragma unroll
        for (int r = 0; r < 8; ++r)
            partial[(size_t)(rowBase + r + hi * 8) * 64 + bc] = rowpart[r];
    }

    // Column-partials (off-diagonal tiles only): fold lane halves (same col,
    // complementary row octets), then reduce the 8 waves through LDS.
    __shared__ float colred[8][128];
    #pragma unroll
    for (int ct = 0; ct < 8; ++ct)
        colpart[ct] += __shfl_xor(colpart[ct], 16, 32);
    if (hi == 0) {
        #pragma unroll
        for (int ct = 0; ct < 8; ++ct)
            colred[wave][ct * 16 + lm] = colpart[ct];
    }
    __syncthreads();
    if (br != bc && threadIdx.x < 128) {
        const int c = threadIdx.x;
        float s = 0.0f;
        #pragma unroll
        for (int w = 0; w < 8; ++w) s += colred[w][c];
        partial[(size_t)(colBase + c) * 64 + br] = s;
    }
}

// ---------------------------------------------------------------------------
// Kernel 3: positive-pair dot products from the hi+lo representation.
// One wave per row j; partner p = (j + 4096) & 8191.
// ---------------------------------------------------------------------------
__global__ __launch_bounds__(256) void posdot_kernel(
    const __bf16* __restrict__ xh, const __bf16* __restrict__ xl,
    float* __restrict__ posdot)
{
    const int j    = blockIdx.x * 8 + (threadIdx.x >> 5);
    const int lane = threadIdx.x & 31;
    const int p    = (j + 4096) & (NROWS - 1);

    const size_t ja = (size_t)j * DDIM + lane * 16;   // 16 contiguous elems/lane
    const size_t pa = (size_t)p * DDIM + lane * 16;

    Frag Ah, Al, Bh, Bl;
    Ah.q[0] = *(const int4*)(xh + ja);     Ah.q[1] = *(const int4*)(xh + ja + 8);
    Al.q[0] = *(const int4*)(xl + ja);     Al.q[1] = *(const int4*)(xl + ja + 8);
    Bh.q[0] = *(const int4*)(xh + pa);     Bh.q[1] = *(const int4*)(xh + pa + 8);
    Bl.q[0] = *(const int4*)(xl + pa);     Bl.q[1] = *(const int4*)(xl + pa + 8);

    float sum = 0.0f;
    #pragma unroll
    for (int e = 0; e < 16; ++e) {
        const float a = (float)Ah.v[e] + (float)Al.v[e];
        const float b = (float)Bh.v[e] + (float)Bl.v[e];
        sum += a * b;
    }
    #pragma unroll
    for (int off = 1; off < 32; off <<= 1)
        sum += __shfl_xor(sum, off, 32);

    if (lane == 0) posdot[j] = sum;
}

// ---------------------------------------------------------------------------
// Kernel 4: deterministic fixed-order final reduction to the scalar loss.
// ---------------------------------------------------------------------------
__global__ __launch_bounds__(256) void finalize_kernel(
    const float* __restrict__ partial, const float* __restrict__ posdot,
    float* __restrict__ out)
{
    const int t = threadIdx.x;
    float local = 0.0f;
    for (int j = t; j < NROWS; j += 256) {
        const float* pp = partial + (size_t)j * 64;
        float rs = 0.0f;
        #pragma unroll
        for (int cb = 0; cb < 64; ++cb) rs += pp[cb];
        local += logf(rs) - INV_T * posdot[j];
    }
    __shared__ float red[256];
    red[t] = local;
    __syncthreads();
    #pragma unroll
    for (int off = 128; off > 0; off >>= 1) {
        if (t < off) red[t] += red[t + off];
        __syncthreads();
    }
    if (t == 0) out[0] = red[0] * (1.0f / (float)NROWS);
}

// ---------------------------------------------------------------------------
extern "C" void kernel_launch(void* const* d_in, const int* in_sizes, int n_in,
                              void* d_out, int out_size, void* d_ws, size_t ws_size,
                              hipStream_t stream)
{
    (void)in_sizes; (void)n_in; (void)out_size; (void)ws_size;

    const float* x   = (const float*)d_in[0];
    float*       out = (float*)d_out;

    char* ws = (char*)d_ws;
    __bf16* xh      = (__bf16*)(ws);
    __bf16* xl      = (__bf16*)(ws + (8ull << 20));               //  8 MB
    float*  posdot  = (float*) (ws + (16ull << 20));              // 16 MB
    float*  partial = (float*) (ws + (16ull << 20) + (64 << 10)); // +64 KB

    rownorm_split_kernel<<<NROWS, 256, 0, stream>>>(x, xh, xl);
    posdot_kernel<<<NROWS / 8, 256, 0, stream>>>(xh, xl, posdot);
    simtile_kernel<<<NTRI, 256, 0, stream>>>(xh, xl, partial);
    finalize_kernel<<<1, 256, 0, stream>>>(partial, posdot, out);
}
